// TransformerBlock_48172353192213
// MI455X (gfx1250) — matrix operand order repaired and checked by reading, not compiled, not run on hardware
//
#include <hip/hip_runtime.h>

// ---------------------------------------------------------------------------
// Types for CDNA5 WMMA (wave32): bf16 A/B fragments (16 x __bf16 per lane),
// f32 C/D fragments (8 x float per lane).
// ---------------------------------------------------------------------------
typedef __attribute__((ext_vector_type(16))) __bf16         bhalf16;
typedef __attribute__((ext_vector_type(8)))  float          floatx8;
typedef __attribute__((ext_vector_type(8)))  unsigned short ushortx8;
typedef __attribute__((ext_vector_type(4)))  unsigned int   uintx4;
typedef __attribute__((ext_vector_type(4)))  int            intx4;
typedef __attribute__((ext_vector_type(8)))  int            intx8;

union Frag16 {          // 16 bf16 = 32 bytes = 8 VGPRs
    bhalf16  v;
    ushortx8 h[2];
};

#define DEV_INLINE __device__ __forceinline__

DEV_INLINE unsigned short f2bf(float f) {   // round-to-nearest-even f32 -> bf16 bits
    unsigned int u = __float_as_uint(f);
    u += 0x7FFFu + ((u >> 16) & 1u);
    return (unsigned short)(u >> 16);
}

DEV_INLINE floatx8 wmma_bf16(const Frag16& a, const Frag16& b, floatx8 c) {
    // D = A(16x32 bf16) * B(32x16 bf16) + C(16x16 f32)
    return __builtin_amdgcn_wmma_f32_16x16x32_bf16(
        /*neg_a=*/false, a.v, /*neg_b=*/false, b.v,
        /*c_mod=*/(short)0, c, /*reuse_a=*/false, /*reuse_b=*/false);
}

// ---------------------------------------------------------------------------
// Tensor Data Mover: 2D tile (16-bit elements) global -> LDS, linear LDS fill.
// D# layout per CDNA5 ISA 8.3/8.4.  Issued by one wave; completion via
// s_wait_tensorcnt + workgroup barrier.
// ---------------------------------------------------------------------------
DEV_INLINE void tdm_load_tile_2d(const unsigned short* gsrc, void* lds_dst,
                                 int tile_w, int tile_h, long long row_stride)
{
    const unsigned long long ga = (unsigned long long)gsrc;
    const unsigned lds_off = (unsigned)(unsigned long long)lds_dst; // LDS aperture: low 32 bits = LDS byte addr
    const unsigned long long st0 = (unsigned long long)row_stride;
    const int td0 = (int)row_stride;     // tensor dim0 (elements) = row stride
    const int td1 = 1 << 22;             // generous row count (no OOB in-use)

    uintx4 g0;
    g0[0] = 1u;                                   // count=1, user descriptor
    g0[1] = (unsigned)__builtin_amdgcn_readfirstlane((int)lds_off);
    g0[2] = (unsigned)__builtin_amdgcn_readfirstlane((int)(ga & 0xFFFFFFFFull));
    g0[3] = (unsigned)__builtin_amdgcn_readfirstlane(
                (int)(((ga >> 32) & 0x1FFFFFFull) | (2u << 30)));  // type=2

    intx8 g1;
    g1[0] = 0x00010000;                                       // data_size = 2B
    g1[1] = __builtin_amdgcn_readfirstlane((int)((td0 & 0xFFFF) << 16));
    g1[2] = __builtin_amdgcn_readfirstlane(
                (int)(((td0 >> 16) & 0xFFFF) | ((td1 & 0xFFFF) << 16)));
    g1[3] = __builtin_amdgcn_readfirstlane(
                (int)(((td1 >> 16) & 0xFFFF) | ((tile_w & 0xFFFF) << 16)));
    g1[4] = __builtin_amdgcn_readfirstlane((int)(tile_h & 0xFFFF)); // tile_dim1
    g1[5] = __builtin_amdgcn_readfirstlane((int)(st0 & 0xFFFFFFFFull));
    g1[6] = __builtin_amdgcn_readfirstlane((int)((st0 >> 32) & 0xFFFFull));
    g1[7] = 0;

    intx4 g2 = {0, 0, 0, 0};
    intx4 g3 = {0, 0, 0, 0};
#if defined(__clang_major__) && (__clang_major__ >= 23)
    intx8 g4 = {0, 0, 0, 0, 0, 0, 0, 0};
    __builtin_amdgcn_tensor_load_to_lds(g0, g1, g2, g3, g4, 0);
#else
    __builtin_amdgcn_tensor_load_to_lds(g0, g1, g2, g3, 0);
#endif
}

DEV_INLINE void tdm_wait() { __builtin_amdgcn_s_wait_tensorcnt(0); }

// Problem constants
static constexpr int Bc = 4, Tc = 1024, Dc = 768, Hc = 12, Kc = 64;
static constexpr int BT = Bc * Tc;           // 4096 rows
static constexpr int HK2 = 2 * Hc * Kc;      // 1536 (q|k packed)

// ---------------------------------------------------------------------------
// LayerNorm (f32 in) -> bf16 out.  One block per row, 256 threads, D = 768.
// ---------------------------------------------------------------------------
__global__ __launch_bounds__(256) void layernorm_to_bf16(
    const float* __restrict__ x, const float* __restrict__ g,
    const float* __restrict__ beta, unsigned short* __restrict__ out)
{
    const int D = Dc;
    const long long row = blockIdx.x;
    const float* xr = x + row * D;
    const int tid = threadIdx.x;

    float v0 = xr[tid], v1 = xr[tid + 256], v2 = xr[tid + 512];
    float s  = v0 + v1 + v2;
    float s2 = v0 * v0 + v1 * v1 + v2 * v2;
    #pragma unroll
    for (int off = 1; off < 32; off <<= 1) {
        s  += __shfl_xor(s,  off, 32);
        s2 += __shfl_xor(s2, off, 32);
    }
    __shared__ float rs[8], rs2[8];
    const int wave = tid >> 5, lane = tid & 31;
    if (lane == 0) { rs[wave] = s; rs2[wave] = s2; }
    __syncthreads();
    if (wave == 0) {
        float a  = (lane < 8) ? rs[lane]  : 0.f;
        float b2 = (lane < 8) ? rs2[lane] : 0.f;
        #pragma unroll
        for (int off = 1; off < 8; off <<= 1) {
            a  += __shfl_xor(a,  off, 32);
            b2 += __shfl_xor(b2, off, 32);
        }
        if (lane == 0) { rs[0] = a; rs2[0] = b2; }
    }
    __syncthreads();
    const float mean = rs[0] / (float)D;
    const float var  = rs2[0] / (float)D - mean * mean;
    const float rstd = rsqrtf(var + 1e-5f);

    unsigned short* orow = out + row * D;
    orow[tid]       = f2bf((v0 - mean) * rstd * g[tid]       + beta[tid]);
    orow[tid + 256] = f2bf((v1 - mean) * rstd * g[tid + 256] + beta[tid + 256]);
    orow[tid + 512] = f2bf((v2 - mean) * rstd * g[tid + 512] + beta[tid + 512]);
}

// ---------------------------------------------------------------------------
// Weight conversions fp32 -> bf16
// ---------------------------------------------------------------------------
__global__ __launch_bounds__(256) void cvt_f32_bf16(
    const float* __restrict__ in, unsigned short* __restrict__ out, long long n)
{
    for (long long i = blockIdx.x * 256LL + threadIdx.x; i < n;
         i += (long long)gridDim.x * 256LL)
        out[i] = f2bf(in[i]);
}

// Wq/Wk (H,D,K) -> packed (D, 2*H*K): cols [0,768)=q heads, [768,1536)=k heads
__global__ __launch_bounds__(256) void pack_qk_bf16(
    const float* __restrict__ Wq, const float* __restrict__ Wk,
    unsigned short* __restrict__ out)
{
    const long long n = (long long)Hc * Dc * Kc;
    for (long long i = blockIdx.x * 256LL + threadIdx.x; i < n;
         i += (long long)gridDim.x * 256LL) {
        int h  = (int)(i / (Dc * Kc));
        int r  = (int)(i % (Dc * Kc));
        int d  = r / Kc;
        int kk = r % Kc;
        out[(long long)d * HK2 + h * Kc + kk]            = f2bf(Wq[i]);
        out[(long long)d * HK2 + Hc * Kc + h * Kc + kk]  = f2bf(Wk[i]);
    }
}

// ---------------------------------------------------------------------------
// Generic bf16 WMMA GEMM: C[M,N] = A[M,K] * B[K,N] (+bias +residual, relu)
// Block tile 128x128, BK=32, 8 waves (4 in M x 2 in N), wave tile 32x64.
// A tile moved by TDM (tensor_load_to_lds); B tile manually transposed to
// [n][k] so both fragment types are two contiguous ds_load_b128 per lane.
// All M,N,K used here are multiples of the tile sizes -> no bounds checks.
// ---------------------------------------------------------------------------
__global__ __launch_bounds__(256) void gemm_bf16_wmma(
    const unsigned short* __restrict__ A, long long sA, int lda,
    const unsigned short* __restrict__ B, long long sB, int ldb,
    int Kd,
    const float* __restrict__ bias,      // [N] or null
    const float* __restrict__ resid,     // [M,ldc] f32 or null
    int relu,
    float* __restrict__ Cf,              // f32 out (or null)
    unsigned short* __restrict__ Cb,     // bf16 out (or null)
    long long sC, int ldc)
{
    constexpr int BM = 128, BN = 128, BK = 32;
    __shared__ unsigned short a_lds[BM * BK];   // [m][k] row-major
    __shared__ unsigned short b_lds[BN * BK];   // [n][k] (transposed)

    const int tid  = threadIdx.x;
    const int lane = tid & 31;
    const int wave = tid >> 5;
    const int wm = wave >> 1;          // 0..3  -> M offset 32*wm
    const int wn = wave & 1;           // 0..1  -> N offset 64*wn
    const int m0 = blockIdx.y * BM;
    const int n0 = blockIdx.x * BN;

    const unsigned short* Ab = A + (long long)blockIdx.z * sA;
    const unsigned short* Bb = B + (long long)blockIdx.z * sB;

    floatx8 acc[2][4];
    #pragma unroll
    for (int mf = 0; mf < 2; ++mf)
        #pragma unroll
        for (int nf = 0; nf < 4; ++nf)
            #pragma unroll
            for (int r = 0; r < 8; ++r) acc[mf][nf][r] = 0.f;

    const int akoff = (lane >> 4) * 8;    // ISA 16-bit A fragment layout
    const int bkoff = (lane >> 4) * 8;    // B fragment: same ISA layout as A

    for (int k0 = 0; k0 < Kd; k0 += BK) {
        __syncthreads();
        if (wave == 0) {   // A tile 128x32 via Tensor Data Mover
            tdm_load_tile_2d(Ab + (long long)m0 * lda + k0, a_lds, BK, BM, lda);
        }
        {   // B tile: 32x128 read row-major, stored transposed [n][k]
            const int kr = tid >> 3, nseg = (tid & 7) * 16;
            const ushortx8* src = reinterpret_cast<const ushortx8*>(
                Bb + (long long)(k0 + kr) * ldb + n0 + nseg);
            ushortx8 e0 = src[0], e1 = src[1];
            #pragma unroll
            for (int j = 0; j < 8; ++j) b_lds[(nseg + j) * BK + kr]     = e0[j];
            #pragma unroll
            for (int j = 0; j < 8; ++j) b_lds[(nseg + 8 + j) * BK + kr] = e1[j];
        }
        if (wave == 0) tdm_wait();
        __syncthreads();

        Frag16 af[2];
        #pragma unroll
        for (int mf = 0; mf < 2; ++mf) {
            const int r = wm * 32 + mf * 16 + (lane & 15);
            af[mf].h[0] = *reinterpret_cast<ushortx8*>(&a_lds[r * BK + akoff]);
            af[mf].h[1] = *reinterpret_cast<ushortx8*>(&a_lds[r * BK + akoff + 16]);
        }
        Frag16 bfr[4];
        #pragma unroll
        for (int nf = 0; nf < 4; ++nf) {
            const int c = wn * 64 + nf * 16 + (lane & 15);
            bfr[nf].h[0] = *reinterpret_cast<ushortx8*>(&b_lds[c * BK + bkoff]);
            bfr[nf].h[1] = *reinterpret_cast<ushortx8*>(&b_lds[c * BK + bkoff + 16]);
        }
        #pragma unroll
        for (int mf = 0; mf < 2; ++mf)
            #pragma unroll
            for (int nf = 0; nf < 4; ++nf)
                acc[mf][nf] = wmma_bf16(af[mf], bfr[nf], acc[mf][nf]);
    }

    // Epilogue: C layout -> lane n = lane&15, row = r + 8*(lane>=16)
    #pragma unroll
    for (int mf = 0; mf < 2; ++mf) {
        #pragma unroll
        for (int nf = 0; nf < 4; ++nf) {
            const int col = n0 + wn * 64 + nf * 16 + (lane & 15);
            const float bv = bias ? bias[col] : 0.f;
            #pragma unroll
            for (int r = 0; r < 8; ++r) {
                const int row = m0 + wm * 32 + mf * 16 + r + ((lane >> 4) << 3);
                float v = acc[mf][nf][r] + bv;
                if (resid) v += resid[(long long)row * ldc + col];
                if (relu)  v = fmaxf(v, 0.f);
                const long long off = (long long)blockIdx.z * sC +
                                      (long long)row * ldc + col;
                if (Cf) Cf[off] = v;
                else    Cb[off] = f2bf(v);
            }
        }
    }
}

// ---------------------------------------------------------------------------
// Causal flash attention. Per-head V is 768 wide, so the grid tiles the
// output dim in 128-col chunks and each block redoes QK^T + online softmax
// for its chunk (scores are <10% of total FLOPs).
// grid = (6 e-chunks, 8 q-tiles of 128 rows, B*H)
// Each of 8 waves owns 16 q-rows; accumulator O = 8 x floatx8 (16x128 f32).
// Q and K tiles moved by TDM; V transposed manually into LDS.
// ---------------------------------------------------------------------------
__global__ __launch_bounds__(256) void attn_flash_bf16(
    const unsigned short* __restrict__ qk,    // [BT][1536] : q|k bf16
    const unsigned short* __restrict__ vbf,   // [H][BT][768] bf16
    unsigned short* __restrict__ attn)        // [BT][9216] bf16 (concat heads)
{
    const int bh = blockIdx.z;
    const int b = bh / Hc, h = bh % Hc;
    const int q0 = blockIdx.y * 128;
    const int e0 = blockIdx.x * 128;

    __shared__ unsigned short q_lds[128 * 64];    // [qrow][kk]
    __shared__ unsigned short k_lds[32 * 64];     // [s][kk]
    __shared__ unsigned short v_lds[128 * 32];    // [e][s] (transposed)
    __shared__ unsigned short p_lds[8 * 16 * 32]; // per-wave 16x32 P tile

    const int tid = threadIdx.x, lane = tid & 31, wave = tid >> 5;

    if (wave == 0) {   // Q tile 128x64 via TDM, row stride 1536
        tdm_load_tile_2d(qk + (long long)(b * Tc + q0) * HK2 + h * Kc,
                         q_lds, Kc, 128, HK2);
        tdm_wait();
    }
    __syncthreads();

    // Per-wave Q fragments (rows wave*16..+16, K-dim 64 = 2 fragments)
    const int akoff = (lane >> 4) * 8;
    const int bkoff = (lane >> 4) * 8;
    Frag16 qf[2];
    {
        const int mrow = wave * 16 + (lane & 15);
        #pragma unroll
        for (int c = 0; c < 2; ++c) {
            qf[c].h[0] = *reinterpret_cast<ushortx8*>(&q_lds[mrow * 64 + c * 32 + akoff]);
            qf[c].h[1] = *reinterpret_cast<ushortx8*>(&q_lds[mrow * 64 + c * 32 + akoff + 16]);
        }
    }

    float m_run[8], l_run[8];
    floatx8 o[8];
    #pragma unroll
    for (int r = 0; r < 8; ++r) { m_run[r] = -__builtin_inff(); l_run[r] = 0.f; }
    #pragma unroll
    for (int e = 0; e < 8; ++e)
        #pragma unroll
        for (int r = 0; r < 8; ++r) o[e][r] = 0.f;

    const int rbase = q0 + wave * 16 + ((lane >> 4) << 3);  // row = rbase + r
    const int s_end = q0 + 128;                              // causal bound

    for (int s0 = 0; s0 < s_end; s0 += 32) {
        __syncthreads();
        if (wave == 0) {   // K tile 32x64 via TDM
            tdm_load_tile_2d(qk + (long long)(b * Tc + s0) * HK2 + Hc * Kc + h * Kc,
                             k_lds, Kc, 32, HK2);
        }
        {   // V chunk: 32 x 128, stored transposed [e][s]
            const int sr = tid >> 3, eseg = (tid & 7) * 16;
            const ushortx8* src = reinterpret_cast<const ushortx8*>(
                vbf + ((long long)h * BT + b * Tc + s0 + sr) * Dc + e0 + eseg);
            ushortx8 v0 = src[0], v1 = src[1];
            #pragma unroll
            for (int j = 0; j < 8; ++j) v_lds[(eseg + j) * 32 + sr]     = v0[j];
            #pragma unroll
            for (int j = 0; j < 8; ++j) v_lds[(eseg + 8 + j) * 32 + sr] = v1[j];
        }
        if (wave == 0) tdm_wait();
        __syncthreads();

        // scores: Q(16x64) @ K^T(64x32) -> 2 n-tiles, 2 K-chunks each
        floatx8 sc[2];
        #pragma unroll
        for (int nt = 0; nt < 2; ++nt) {
            #pragma unroll
            for (int r = 0; r < 8; ++r) sc[nt][r] = 0.f;
            #pragma unroll
            for (int c = 0; c < 2; ++c) {
                Frag16 kf;
                const int srow = nt * 16 + (lane & 15);
                kf.h[0] = *reinterpret_cast<ushortx8*>(&k_lds[srow * 64 + c * 32 + bkoff]);
                kf.h[1] = *reinterpret_cast<ushortx8*>(&k_lds[srow * 64 + c * 32 + bkoff + 16]);
                sc[nt] = wmma_bf16(qf[c], kf, sc[nt]);
            }
        }

        // scale + causal mask + online softmax (rows in VGPR idx, cols in lanes)
        const float scale = 0.125f;   // 1/sqrt(64)
        float mt[8];
        #pragma unroll
        for (int r = 0; r < 8; ++r) {
            const int row = rbase + r;
            #pragma unroll
            for (int nt = 0; nt < 2; ++nt) {
                const int col = s0 + nt * 16 + (lane & 15);
                float sv = sc[nt][r] * scale;
                if (col > row) sv = -__builtin_inff();
                sc[nt][r] = sv;
            }
            mt[r] = fmaxf(sc[0][r], sc[1][r]);
        }
        #pragma unroll
        for (int r = 0; r < 8; ++r)
            #pragma unroll
            for (int off = 1; off < 16; off <<= 1)
                mt[r] = fmaxf(mt[r], __shfl_xor(mt[r], off, 32));

        #pragma unroll
        for (int r = 0; r < 8; ++r) {
            const float m_new = fmaxf(m_run[r], mt[r]);
            const float alpha = expf(m_run[r] - m_new);   // 0 on first tile
            sc[0][r] = expf(sc[0][r] - m_new);
            sc[1][r] = expf(sc[1][r] - m_new);
            float rs = sc[0][r] + sc[1][r];
            #pragma unroll
            for (int off = 1; off < 16; off <<= 1) rs += __shfl_xor(rs, off, 32);
            l_run[r] = l_run[r] * alpha + rs;
            m_run[r] = m_new;
            #pragma unroll
            for (int e = 0; e < 8; ++e) o[e][r] *= alpha;
        }

        // P (C layout) -> LDS -> A-fragment layout
        #pragma unroll
        for (int r = 0; r < 8; ++r) {
            const int prow = r + ((lane >> 4) << 3);
            #pragma unroll
            for (int nt = 0; nt < 2; ++nt)
                p_lds[wave * 512 + prow * 32 + nt * 16 + (lane & 15)] = f2bf(sc[nt][r]);
        }
        __syncthreads();

        Frag16 pf;
        pf.h[0] = *reinterpret_cast<ushortx8*>(&p_lds[wave * 512 + (lane & 15) * 32 + akoff]);
        pf.h[1] = *reinterpret_cast<ushortx8*>(&p_lds[wave * 512 + (lane & 15) * 32 + akoff + 16]);

        #pragma unroll
        for (int et = 0; et < 8; ++et) {
            Frag16 vf;
            const int ecol = et * 16 + (lane & 15);
            vf.h[0] = *reinterpret_cast<ushortx8*>(&v_lds[ecol * 32 + bkoff]);
            vf.h[1] = *reinterpret_cast<ushortx8*>(&v_lds[ecol * 32 + bkoff + 16]);
            o[et] = wmma_bf16(pf, vf, o[et]);
        }
    }

    // finalize: O /= l ; store bf16 into concat-head layout [BT][9216]
    #pragma unroll
    for (int r = 0; r < 8; ++r) {
        const float inv = 1.0f / l_run[r];
        const int row = rbase + r;
        #pragma unroll
        for (int et = 0; et < 8; ++et) {
            const int col = e0 + et * 16 + (lane & 15);
            attn[(long long)(b * Tc + row) * (Hc * Dc) + h * Dc + col] =
                f2bf(o[et][r] * inv);
        }
    }
}

// ---------------------------------------------------------------------------
// Host-side pipeline
// ---------------------------------------------------------------------------
extern "C" void kernel_launch(void* const* d_in, const int* in_sizes, int n_in,
                              void* d_out, int out_size, void* d_ws, size_t ws_size,
                              hipStream_t stream) {
    (void)in_sizes; (void)n_in; (void)out_size; (void)ws_size;
    const float* x     = (const float*)d_in[0];
    const float* ln1_g = (const float*)d_in[1];
    const float* ln1_b = (const float*)d_in[2];
    const float* ln2_g = (const float*)d_in[3];
    const float* ln2_b = (const float*)d_in[4];
    const float* Wq    = (const float*)d_in[5];
    const float* Wk    = (const float*)d_in[6];
    const float* Wv    = (const float*)d_in[7];
    const float* Wproj = (const float*)d_in[8];
    const float* bproj = (const float*)d_in[9];
    const float* W1    = (const float*)d_in[10];
    const float* b1    = (const float*)d_in[11];
    const float* W2    = (const float*)d_in[12];
    const float* b2    = (const float*)d_in[13];
    float* out = (float*)d_out;

    char* ws = (char*)d_ws;
    size_t cur = 0;
    auto alloc = [&](size_t bytes) -> void* {
        cur = (cur + 255) & ~(size_t)255;
        void* p = ws + cur;
        cur += bytes;
        return p;
    };

    unsigned short* h1_bf   = (unsigned short*)alloc((size_t)BT * Dc * 2);
    unsigned short* wqk_bf  = (unsigned short*)alloc((size_t)Dc * HK2 * 2);
    unsigned short* wv_bf   = (unsigned short*)alloc((size_t)Hc * Dc * Dc * 2);
    unsigned short* wpr_bf  = (unsigned short*)alloc((size_t)Hc * Dc * Dc * 2);
    unsigned short* w1_bf   = (unsigned short*)alloc((size_t)Dc * 4 * Dc * 2);
    unsigned short* w2_bf   = (unsigned short*)alloc((size_t)4 * Dc * Dc * 2);
    unsigned short* qk_bf   = (unsigned short*)alloc((size_t)BT * HK2 * 2);
    unsigned short* v_bf    = (unsigned short*)alloc((size_t)Hc * BT * Dc * 2);
    unsigned short* attn_bf = (unsigned short*)alloc((size_t)BT * Hc * Dc * 2);
    float*          x2      = (float*)alloc((size_t)BT * Dc * 4);
    unsigned short* h2_bf   = (unsigned short*)alloc((size_t)BT * Dc * 2);
    unsigned short* a1_bf   = (unsigned short*)alloc((size_t)BT * 4 * Dc * 2);

    // 1. h1 = LN1(x) -> bf16
    layernorm_to_bf16<<<BT, 256, 0, stream>>>(x, ln1_g, ln1_b, h1_bf);

    // 2. weight conversion
    pack_qk_bf16<<<2304, 256, 0, stream>>>(Wq, Wk, wqk_bf);
    cvt_f32_bf16<<<2048, 256, 0, stream>>>(Wv,    wv_bf,  (long long)Hc * Dc * Dc);
    cvt_f32_bf16<<<2048, 256, 0, stream>>>(Wproj, wpr_bf, (long long)Hc * Dc * Dc);
    cvt_f32_bf16<<<1024, 256, 0, stream>>>(W1,    w1_bf,  (long long)Dc * 4 * Dc);
    cvt_f32_bf16<<<1024, 256, 0, stream>>>(W2,    w2_bf,  (long long)4 * Dc * Dc);

    // 3. q|k = h1 @ [Wq|Wk]   (4096 x 768 x 1536) -> bf16
    gemm_bf16_wmma<<<dim3(HK2 / 128, BT / 128, 1), 256, 0, stream>>>(
        h1_bf, 0, Dc, wqk_bf, 0, HK2, Dc,
        nullptr, nullptr, 0, nullptr, qk_bf, 0, HK2);

    // 4. v_h = h1 @ Wv[h], batched over 12 heads -> bf16
    gemm_bf16_wmma<<<dim3(Dc / 128, BT / 128, Hc), 256, 0, stream>>>(
        h1_bf, 0, Dc, wv_bf, (long long)Dc * Dc, Dc, Dc,
        nullptr, nullptr, 0, nullptr, v_bf, (long long)BT * Dc, Dc);

    // 5. causal flash attention -> attn_bf (concat-head layout [BT][9216])
    attn_flash_bf16<<<dim3(Dc / 128, Tc / 128, Bc * Hc), 256, 0, stream>>>(
        qk_bf, v_bf, attn_bf);

    // 6. x2 = x + attn @ Wproj + bproj   (4096 x 9216 x 768) -> f32
    gemm_bf16_wmma<<<dim3(Dc / 128, BT / 128, 1), 256, 0, stream>>>(
        attn_bf, 0, Hc * Dc, wpr_bf, 0, Dc, Hc * Dc,
        bproj, x, 0, x2, nullptr, 0, Dc);

    // 7. h2 = LN2(x2) -> bf16
    layernorm_to_bf16<<<BT, 256, 0, stream>>>(x2, ln2_g, ln2_b, h2_bf);

    // 8. a1 = relu(h2 @ W1 + b1)  (4096 x 768 x 3072) -> bf16
    gemm_bf16_wmma<<<dim3(4 * Dc / 128, BT / 128, 1), 256, 0, stream>>>(
        h2_bf, 0, Dc, w1_bf, 0, 4 * Dc, Dc,
        b1, nullptr, 1, nullptr, a1_bf, 0, 4 * Dc);

    // 9. out = x2 + a1 @ W2 + b2  (4096 x 3072 x 768) -> f32
    gemm_bf16_wmma<<<dim3(Dc / 128, BT / 128, 1), 256, 0, stream>>>(
        a1_bf, 0, 4 * Dc, w2_bf, 0, Dc, 4 * Dc,
        b2, x2, 0, out, nullptr, 0, Dc);
}
